// PointCloudMessagePassing_46514495815850
// MI455X (gfx1250) — compile-verified, matrix operand order checked by reading, not run on hardware
//
#include <hip/hip_runtime.h>
#include <hip/hip_bf16.h>
#include <stdint.h>

// ---------------- problem constants ----------------
#define BB      2
#define N0C     2000
#define FEAT    256
#define NCLS    9
#define NNC     (N0C + NCLS)     // 2009
#define DD      128              // per-layer dim
#define HHH     4                // heads
#define DH      32               // head dim
#define KCAP    100              // neighbor cap
#define MMC     250
#define HID     512
#define LP      112              // padded neighbor rows (7*16)
#define LDX     136              // f16 LDS row stride (halves)

typedef __attribute__((ext_vector_type(16))) _Float16 v16h;
typedef __attribute__((ext_vector_type(8)))  float    v8f;

// ---------------- WMMA helpers (CDNA5 wave32) ----------------
static __device__ __forceinline__ v8f wmma16(v16h a, v16h b, v8f c) {
  // v_wmma_f32_16x16x32_f16  D = A(16x32) * B(32x16) + C(16x16)
  return __builtin_amdgcn_wmma_f32_16x16x32_f16(false, a, false, b, (short)0, c,
                                                false, false);
}

// A-matrix loader (16x32 f16): lane l%16 = row, lane-hi selects K 0..7/16..23 vs 8..15/24..31
static __device__ __forceinline__ v16h loadA(const _Float16* base, int ld, int r0, int k0) {
  int l = threadIdx.x & 31;
  const _Float16* p = base + (size_t)(r0 + (l & 15)) * ld + k0 + ((l >> 4) << 3);
  v16h a;
#pragma unroll
  for (int i = 0; i < 8; ++i) { a[i] = p[i]; a[i + 8] = p[16 + i]; }
  return a;
}
// B operand from row-major "weight" W[N][K]:  B[k][n] = W[n0+n][k0+k]  (lane = column)
static __device__ __forceinline__ v16h loadBw(const _Float16* base, int ld, int n0, int k0) {
  return loadA(base, ld, n0, k0);
}
// B operand from row-major [K][N] matrix:  B[k][n] = base[(k0+k)*ld + c0+n]
static __device__ __forceinline__ v16h loadBk(const _Float16* base, int ld, int k0, int c0) {
  int l = threadIdx.x & 31;
  const _Float16* p = base + (size_t)(k0 + ((l >> 4) << 3)) * ld + c0 + (l & 15);
  v16h b;
#pragma unroll
  for (int i = 0; i < 8; ++i) { b[i] = p[(size_t)i * ld]; b[i + 8] = p[(size_t)(16 + i) * ld]; }
  return b;
}
// C/D store: VGPR r -> row r0 + r + 8*(lane>=16), col = c0 + lane&15
static __device__ __forceinline__ void storeC16(_Float16* base, int ld, int r0, int c0, v8f c) {
  int l = threadIdx.x & 31;
  _Float16* p = base + (size_t)(r0 + ((l >> 4) << 3)) * ld + c0 + (l & 15);
#pragma unroll
  for (int r = 0; r < 8; ++r) p[(size_t)r * ld] = (_Float16)c[r];
}
static __device__ __forceinline__ void storeC16s(_Float16* base, int ld, int r0, int c0, v8f c, float s) {
  int l = threadIdx.x & 31;
  _Float16* p = base + (size_t)(r0 + ((l >> 4) << 3)) * ld + c0 + (l & 15);
#pragma unroll
  for (int r = 0; r < 8; ++r) p[(size_t)r * ld] = (_Float16)(c[r] * s);
}

// ---------------- small utility kernels ----------------
__global__ void zero_u32(unsigned* __restrict__ p, int n) {
  int t = blockIdx.x * blockDim.x + threadIdx.x;
  if (t < n) p[t] = 0u;
}
__global__ void cvt_f16(const float* __restrict__ s, _Float16* __restrict__ d, int n) {
  int t = blockIdx.x * blockDim.x + threadIdx.x;
  if (t < n) d[t] = (_Float16)s[t];
}
__global__ void corr_kernel(const int* __restrict__ corr, unsigned* __restrict__ cmask) {
  int t = blockIdx.x * blockDim.x + threadIdx.x;
  if (t >= BB * 8 * MMC) return;
  int b = t / (8 * MMC);
  int s = (t / MMC) & 7;
  int r = corr[t];
  atomicOr(&cmask[b * N0C + r], 1u << s);
}

// feats = biochem @ w_in^T + b_in ; cls rows appended.  Stores f16 and relu-f16 copies.
__global__ __launch_bounds__(128)
void feats_kernel(const float* __restrict__ bio, const float* __restrict__ cls,
                  const float* __restrict__ w_in, const float* __restrict__ b_in,
                  _Float16* __restrict__ feats16, _Float16* __restrict__ featsR16) {
  __shared__ float xrow[FEAT];
  int row = blockIdx.x;                // b*NN + n
  int n = row % NNC;
  int d = threadIdx.x;                 // 128 threads
  float v;
  if (n < N0C) {                       // uniform branch per block
    int b = row / NNC;
    const float* src = bio + ((size_t)b * N0C + n) * FEAT;
    xrow[d] = src[d];
    xrow[d + 128] = src[d + 128];
    __syncthreads();
    const float* w = w_in + (size_t)d * FEAT;
    float s = b_in[d];
    for (int k = 0; k < FEAT; ++k) s = fmaf(xrow[k], w[k], s);
    v = s;
  } else {
    v = cls[(size_t)(n - N0C) * DD + d];
  }
  size_t o = (size_t)row * DD + d;
  feats16[o]  = (_Float16)v;
  featsR16[o] = (_Float16)fmaxf(v, 0.f);
}

// global max pairwise distance (reference formula: sq_i + sq_j - 2 x.x, sqrt(max(.,0)))
__global__ void maxd_kernel(const float* __restrict__ surf, unsigned* __restrict__ maxd_u) {
  int t = blockIdx.x * blockDim.x + threadIdx.x;
  if (t >= BB * N0C) return;
  int b = t / N0C, n = t % N0C;
  const float* base = surf + (size_t)b * N0C * 3;
  float xn = base[n * 3], yn = base[n * 3 + 1], zn = base[n * 3 + 2];
  float sqn = xn * xn + yn * yn + zn * zn;
  float mx = 0.f;
  for (int j = 0; j < N0C; ++j) {
    float xj = base[j * 3], yj = base[j * 3 + 1], zj = base[j * 3 + 2];
    float sqj = xj * xj + yj * yj + zj * zj;
    float d2 = sqn + sqj - 2.f * (xn * xj + yn * yj + zn * zj);
    mx = fmaxf(mx, sqrtf(fmaxf(d2, 0.f)));
  }
  atomicMax(maxd_u, __float_as_uint(mx));   // nonneg floats: uint order == float order
}

// neighbor selection: first KCAP ascending-index cols with dist<thr_i, then col N0 (always),
// then cls cols N0+1+s for subareas containing this node.  One wave32 per (b,n).
__global__ __launch_bounds__(128)
void select_kernel(const float* __restrict__ surf, const unsigned* __restrict__ cmask,
                   const unsigned* __restrict__ maxd_u,
                   int* __restrict__ idxbuf, int* __restrict__ cnts) {
  int lane = threadIdx.x & 31;
  int gid = blockIdx.x * 4 + (threadIdx.x >> 5);
  if (gid >= BB * NNC) return;
  int b = gid / NNC, n = gid % NNC;
  int* out = idxbuf + (size_t)gid * 4 * KCAP;
  if (n >= N0C) {                       // cls rows: only self
    if (lane == 0)
      for (int i = 0; i < 4; ++i) { out[i * KCAP] = n; cnts[gid * 4 + i] = 1; }
    return;
  }
  float maxd = __uint_as_float(*maxd_u);
  float thr[4];
#pragma unroll
  for (int i = 0; i < 4; ++i) thr[i] = maxd * (0.0125f * (float)(i + 1)); // maxd/20*((i+1)/4)
  const float* base = surf + (size_t)b * N0C * 3;
  float xn = base[n * 3], yn = base[n * 3 + 1], zn = base[n * 3 + 2];
  float sqn = xn * xn + yn * yn + zn * zn;
  int cnt[4] = {0, 0, 0, 0};
  unsigned long long lowm = (1ull << lane) - 1ull;
  for (int j0 = 0; j0 < N0C; j0 += 32) {
    int j = j0 + lane;
    float d = 3.0e38f;
    if (j < N0C) {
      float xj = base[j * 3], yj = base[j * 3 + 1], zj = base[j * 3 + 2];
      float sqj = xj * xj + yj * yj + zj * zj;
      float d2 = sqn + sqj - 2.f * (xn * xj + yn * yj + zn * zj);
      d = sqrtf(fmaxf(d2, 0.f));
    }
#pragma unroll
    for (int i = 0; i < 4; ++i) {
      bool p = d < thr[i];
      unsigned long long mb = __ballot(p);
      int pos = cnt[i] + __popcll(mb & lowm);
      if (p && pos < KCAP) out[i * KCAP + pos] = j;
      cnt[i] += __popcll(mb);
      if (cnt[i] > KCAP) cnt[i] = KCAP;
    }
    if (cnt[0] >= KCAP) break;          // thr[0] smallest -> all scales full
  }
  if (lane == 0) {
    unsigned cm = cmask[b * N0C + n];
    for (int i = 0; i < 4; ++i) {
      int c = cnt[i];
      if (c < KCAP) out[i * KCAP + c++] = N0C;        // global cls col, dist 0
      for (int s = 0; s < 8; ++s)
        if (((cm >> s) & 1u) && c < KCAP) out[i * KCAP + c++] = N0C + 1 + s;
      cnts[gid * 4 + i] = c;
    }
  }
}

// ---------------- per-(node,scale) attention tile (WMMA heavy) ----------------
// dynamic LDS layout (bytes) -- 157632 total, so 2 workgroups fit per WGP:
//  X/O : 0       112*136*2 = 30464   (O reuses X after QKV)
//  Q   : 30464
//  K   : 60928
//  V   : 91392   128*136*2 = 34816   (rows 112..127 zeroed for B-side reads)
//  SP  : 126208  112*136*2           (f16 scores, then in-place softmax probs)
//  pool: 156672  512
//  idx : 157184  448
#define SH_Q   30464
#define SH_K   60928
#define SH_V   91392
#define SH_SP  126208
#define SH_PL  156672
#define SH_IX  157184
#define SH_TOT 157632

__global__ __launch_bounds__(128)
void attn_kernel(const _Float16* __restrict__ feats16,
                 const _Float16* __restrict__ ipw16, const float* __restrict__ ipb,
                 const _Float16* __restrict__ outw16, const float* __restrict__ outb,
                 const int* __restrict__ idxbuf, const int* __restrict__ cnts,
                 float* __restrict__ combined) {
  extern __shared__ char smem[];
  _Float16* Xs = (_Float16*)smem;           // also O
  _Float16* Qs = (_Float16*)(smem + SH_Q);
  _Float16* Ks = (_Float16*)(smem + SH_K);
  _Float16* Vs = (_Float16*)(smem + SH_V);
  _Float16* SP = (_Float16*)(smem + SH_SP); // scores -> probabilities (row-in-place)
  float*    pooled = (float*)(smem + SH_PL);
  int*      idxl = (int*)(smem + SH_IX);
  _Float16* Os = Xs;

  int tid = threadIdx.x, lane = tid & 31, wid = tid >> 5;
  int blk = blockIdx.x;
  int sc4 = blk & 3;
  int gid = blk >> 2;                       // b*NN + n
  int b = gid / NNC;
  int cnt = cnts[gid * 4 + sc4];
  if (cnt < 1) cnt = 1;
  if (cnt > KCAP) cnt = KCAP;

  if (tid < LP) {
    int v = 0;
    if (tid < cnt) v = idxbuf[((size_t)gid * 4 + sc4) * KCAP + tid];
    idxl[tid] = v;
  }
  if (tid < DD) pooled[tid] = 0.f;
  __syncthreads();

  // gather neighbor features (zeros beyond cnt); zero X pad cols; zero V rows 112..127
  for (int l = wid; l < LP; l += 4) {
    uint2 v = make_uint2(0u, 0u);
    if (l < cnt) {
      const uint2* src = (const uint2*)(feats16 + ((size_t)b * NNC + idxl[l]) * DD);
      v = src[lane];
    }
    *(uint2*)(Xs + (size_t)l * LDX + lane * 4) = v;
    if (lane < 4) ((unsigned*)(Xs + (size_t)l * LDX + DD))[lane] = 0u;
  }
  for (int l = LP + wid; l < 128; l += 4)
    *(uint2*)(Vs + (size_t)l * LDX + lane * 4) = make_uint2(0u, 0u);
  __syncthreads();

  // QKV = X @ in_proj_w^T + in_proj_b   (168 C-tiles split over 4 waves)
  for (int t = wid; t < 168; t += 4) {
    int which = t / 56, r = t % 56, mt = r >> 3, nt = r & 7;
    float bv = ipb[which * DD + nt * 16 + (lane & 15)];
    v8f c;
#pragma unroll
    for (int i = 0; i < 8; ++i) c[i] = bv;
#pragma unroll
    for (int k0 = 0; k0 < DD; k0 += 32) {
      v16h a  = loadA(Xs, LDX, mt * 16, k0);
      v16h bm = loadBw(ipw16, DD, which * DD + nt * 16, k0);
      c = wmma16(a, bm, c);
    }
    _Float16* dst = (which == 0) ? Qs : (which == 1) ? Ks : Vs;
    storeC16(dst, LDX, mt * 16, nt * 16, c);
  }
  __syncthreads();

  const float scl = 0.17677669529663687f;   // 1/sqrt(32)
  for (int h = 0; h < HHH; ++h) {
    // SP = (Qh @ Kh^T) * scale, stored f16  (49 tiles, single 32-wide k-step per head)
    for (int t = wid; t < 49; t += 4) {
      int mt = t / 7, nk = t % 7;
      v8f c;
#pragma unroll
      for (int i = 0; i < 8; ++i) c[i] = 0.f;
      v16h a  = loadA(Qs, LDX, mt * 16, h * DH);
      v16h bm = loadBw(Ks, LDX, nk * 16, h * DH);
      c = wmma16(a, bm, c);
      storeC16s(SP, LDX, mt * 16, nk * 16, c, scl);
    }
    __syncthreads();
    // row softmax over valid keys only; in-place (each row owned by one thread)
    if (tid < LP) {
      _Float16* srow = SP + (size_t)tid * LDX;
      if (tid < cnt) {
        float mx = -3.0e38f;
        for (int j = 0; j < cnt; ++j) mx = fmaxf(mx, (float)srow[j]);
        float sum = 0.f;
        for (int j = 0; j < cnt; ++j) {
          float e = __expf((float)srow[j] - mx);
          sum += e;
          srow[j] = (_Float16)e;
        }
        float inv = 1.f / sum;
        for (int j = 0; j < LDX; ++j)
          srow[j] = (j < cnt) ? (_Float16)((float)srow[j] * inv) : (_Float16)0.f;
      } else {
        for (int j = 0; j < LDX; ++j) srow[j] = (_Float16)0.f;
      }
    }
    __syncthreads();
    // O[:, h*32 : h*32+32] = P @ Vh   (14 tiles x 4 k-steps)
    for (int t = wid; t < 14; t += 4) {
      int mt = t >> 1, nt = t & 1;
      v8f c;
#pragma unroll
      for (int i = 0; i < 8; ++i) c[i] = 0.f;
#pragma unroll
      for (int k0 = 0; k0 < 128; k0 += 32) {
        v16h a  = loadA(SP, LDX, mt * 16, k0);
        v16h bm = loadBk(Vs, LDX, k0, h * DH + nt * 16);
        c = wmma16(a, bm, c);
      }
      storeC16(Os, LDX, mt * 16, h * DH + nt * 16, c);
    }
    __syncthreads();
  }

  // out-proj + masked mean pool:  pooled_d = sum_{l<cnt}(O @ out_w^T)[l,d]
  for (int t = wid; t < 56; t += 4) {
    int mt = t >> 3, nt = t & 7;
    v8f c;
#pragma unroll
    for (int i = 0; i < 8; ++i) c[i] = 0.f;
#pragma unroll
    for (int k0 = 0; k0 < DD; k0 += 32) {
      v16h a  = loadA(Os, LDX, mt * 16, k0);
      v16h bm = loadBw(outw16, DD, nt * 16, k0);
      c = wmma16(a, bm, c);
    }
    int col = nt * 16 + (lane & 15);
    int hi = lane >> 4;
    float s = 0.f;
#pragma unroll
    for (int r = 0; r < 8; ++r) {
      int row = mt * 16 + r + (hi << 3);
      if (row < cnt) s += c[r];
    }
    atomicAdd(&pooled[col], s);
  }
  __syncthreads();
  if (tid < DD) {
    float v = pooled[tid] / (float)cnt + outb[tid];
    combined[((size_t)gid * 4 + sc4) * DD + tid] = v;
  }
}

// ---------------- generic WMMA GEMM: C[M][N] = A16[M][K] @ W16[N][K]^T + bias (+C) ---------
__global__ __launch_bounds__(128)
void wmma_gemm_nt(const _Float16* __restrict__ A, const _Float16* __restrict__ W,
                  const float* __restrict__ bias, float* __restrict__ C,
                  int Mrows, int Ncols, int Kdim, int addC) {
  int lane = threadIdx.x & 31;
  int tile = blockIdx.x * 4 + (threadIdx.x >> 5);
  int ntiles = Ncols >> 4;
  int mtiles = (Mrows + 15) >> 4;
  if (tile >= mtiles * ntiles) return;
  int mt = tile / ntiles, nt = tile % ntiles;
  int m = lane & 15, hi = lane >> 4;
  int arow = mt * 16 + m;
  bool aok = arow < Mrows;
  const _Float16* ap = A + (size_t)arow * Kdim + (hi << 3);
  const _Float16* wp = W + (size_t)(nt * 16 + m) * Kdim + (hi << 3);
  float bv = bias[nt * 16 + m];
  v8f c;
#pragma unroll
  for (int i = 0; i < 8; ++i) c[i] = bv;
  if (addC) {
#pragma unroll
    for (int r = 0; r < 8; ++r) {
      int rr = mt * 16 + r + (hi << 3);
      if (rr < Mrows) c[r] += C[(size_t)rr * Ncols + nt * 16 + m];
    }
  }
  for (int k0 = 0; k0 < Kdim; k0 += 32) {
    v16h a;
#pragma unroll
    for (int i = 0; i < 16; ++i) a[i] = (_Float16)0.f;
    if (aok) {
#pragma unroll
      for (int i = 0; i < 8; ++i) { a[i] = ap[k0 + i]; a[i + 8] = ap[k0 + 16 + i]; }
    }
    v16h bm;
#pragma unroll
    for (int i = 0; i < 8; ++i) { bm[i] = wp[k0 + i]; bm[i + 8] = wp[k0 + 16 + i]; }
    c = wmma16(a, bm, c);
  }
#pragma unroll
  for (int r = 0; r < 8; ++r) {
    int rr = mt * 16 + r + (hi << 3);
    if (rr < Mrows) C[(size_t)rr * Ncols + nt * 16 + m] = c[r];
  }
}

// ---------------- host launcher ----------------
extern "C" void kernel_launch(void* const* d_in, const int* in_sizes, int n_in,
                              void* d_out, int out_size, void* d_ws, size_t ws_size,
                              hipStream_t stream) {
  (void)in_sizes; (void)n_in; (void)out_size; (void)ws_size;
  const float* surf = (const float*)d_in[0];
  const float* bio  = (const float*)d_in[1];
  const int*   corr = (const int*)  d_in[2];
  const float* cls  = (const float*)d_in[3];
  const float* w_in = (const float*)d_in[4];
  const float* b_in = (const float*)d_in[5];
  const float* ipw  = (const float*)d_in[6];
  const float* ipb  = (const float*)d_in[7];
  const float* outw = (const float*)d_in[8];
  const float* outb = (const float*)d_in[9];
  const float* wres = (const float*)d_in[10];
  const float* bres = (const float*)d_in[11];
  const float* wfc  = (const float*)d_in[12];
  const float* bfc  = (const float*)d_in[13];
  float* out = (float*)d_out;

  char* ws = (char*)d_ws;
  size_t off = 0;
  auto alloc = [&](size_t bytes) -> char* {
    char* p = ws + off;
    off = (off + bytes + 255) & ~(size_t)255;
    return p;
  };
  _Float16* feats16  = (_Float16*)alloc((size_t)BB * NNC * DD * 2);
  _Float16* featsR16 = (_Float16*)alloc((size_t)BB * NNC * DD * 2);
  float*    combined = (float*)   alloc((size_t)BB * NNC * HID * 4);
  _Float16* comb16   = (_Float16*)alloc((size_t)BB * NNC * HID * 2);
  _Float16* ipw16    = (_Float16*)alloc((size_t)3 * DD * DD * 2);
  _Float16* outw16   = (_Float16*)alloc((size_t)DD * DD * 2);
  _Float16* wres16   = (_Float16*)alloc((size_t)HID * DD * 2);
  _Float16* wfc16    = (_Float16*)alloc((size_t)HID * HID * 2);
  int*      idxbuf   = (int*)     alloc((size_t)BB * NNC * 4 * KCAP * 4);
  int*      cntbuf   = (int*)     alloc((size_t)BB * NNC * 4 * 4);
  unsigned* cmask    = (unsigned*)alloc((size_t)BB * N0C * 4);
  unsigned* maxdu    = (unsigned*)alloc(256);

  // init
  zero_u32<<<(BB * N0C + 255) / 256, 256, 0, stream>>>(cmask, BB * N0C);
  zero_u32<<<1, 32, 0, stream>>>(maxdu, 1);
  // weight conversions to f16
  cvt_f16<<<(3 * DD * DD + 255) / 256, 256, 0, stream>>>(ipw, ipw16, 3 * DD * DD);
  cvt_f16<<<(DD * DD + 255) / 256, 256, 0, stream>>>(outw, outw16, DD * DD);
  cvt_f16<<<(HID * DD + 255) / 256, 256, 0, stream>>>(wres, wres16, HID * DD);
  cvt_f16<<<(HID * HID + 255) / 256, 256, 0, stream>>>(wfc, wfc16, HID * HID);
  // pipeline
  corr_kernel<<<(BB * 8 * MMC + 255) / 256, 256, 0, stream>>>(corr, cmask);
  feats_kernel<<<BB * NNC, 128, 0, stream>>>(bio, cls, w_in, b_in, feats16, featsR16);
  maxd_kernel<<<(BB * N0C + 255) / 256, 256, 0, stream>>>(surf, maxdu);
  select_kernel<<<(BB * NNC + 3) / 4, 128, 0, stream>>>(surf, cmask, maxdu, idxbuf, cntbuf);
  attn_kernel<<<BB * NNC * 4, 128, SH_TOT, stream>>>(feats16, ipw16, ipb, outw16, outb,
                                                     idxbuf, cntbuf, combined);
  // combined += relu(feats) @ w_res^T + b_res
  {
    int mtiles = (BB * NNC + 15) / 16, ntiles = HID / 16;
    int blocks = (mtiles * ntiles + 3) / 4;
    wmma_gemm_nt<<<blocks, 128, 0, stream>>>(featsR16, wres16, bres, combined,
                                             BB * NNC, HID, DD, 1);
  }
  cvt_f16<<<(BB * NNC * HID + 255) / 256, 256, 0, stream>>>(combined, comb16, BB * NNC * HID);
  // out = combined @ w_fc^T + b_fc
  {
    int mtiles = (BB * NNC + 15) / 16, ntiles = HID / 16;
    int blocks = (mtiles * ntiles + 3) / 4;
    wmma_gemm_nt<<<blocks, 128, 0, stream>>>(comb16, wfc16, bfc, out,
                                             BB * NNC, HID, HID, 0);
  }
}
